// HeteroGnnPool_29025388986926
// MI455X (gfx1250) — matrix operand order repaired and checked by reading, not compiled, not run on hardware
//
#include <hip/hip_runtime.h>

#define N_NODES 50000
#define N_EDGES 1600000
#define N_GRAPHS 64
#define DIN 64
#define DH 128

typedef __attribute__((ext_vector_type(16))) _Float16 v16h;
typedef __attribute__((ext_vector_type(8)))  _Float16 v8h;
typedef __attribute__((ext_vector_type(8)))  float    v8f;

// ---------------- utility kernels ----------------

__global__ void k_fill(float* __restrict__ p, float v, int n) {
    int i = blockIdx.x * blockDim.x + threadIdx.x;
    if (i < n) p[i] = v;
}

__global__ void k_deg_count(const int* __restrict__ dst, float* __restrict__ deg, int e) {
    int i = blockIdx.x * blockDim.x + threadIdx.x;
    if (i < e) atomicAdd(&deg[dst[i]], 1.0f);
}

__global__ void k_rsqrt(float* __restrict__ p, int n) {
    int i = blockIdx.x * blockDim.x + threadIdx.x;
    if (i < n) p[i] = rsqrtf(p[i]);
}

__global__ void k_f32_to_f16(const float* __restrict__ in, _Float16* __restrict__ out, int n) {
    int i = blockIdx.x * blockDim.x + threadIdx.x;
    if (i < n) out[i] = (_Float16)in[i];
}

// W[K x DH] row-major f32  ->  WT[DH x K] row-major f16  (so a column of W is contiguous)
__global__ void k_wt(const float* __restrict__ W, _Float16* __restrict__ WT, int K) {
    int t = blockIdx.x * blockDim.x + threadIdx.x;
    if (t < K * DH) {
        int n = t / K, k = t % K;
        WT[t] = (_Float16)W[(size_t)k * DH + n];
    }
}

// ---------------- WMMA GEMM: C[N x 128] = A[N x K](f16) * W (given as WT[128 x K] f16) ----------------
// One 16x16 output tile per wave; 8 waves/block cover the 128 output columns.
// N is a multiple of 16 (50000 = 3125*16) so EXEC stays all-ones (WMMA requirement).
// Dual accumulators break the WMMA->WMMA RAW chain (avoids the 1-NOP+4-coexec hazard stall).
template <int K>
__global__ __launch_bounds__(256)
void k_gemm(const _Float16* __restrict__ A, const _Float16* __restrict__ WT,
            float* __restrict__ C) {
    const int lane = threadIdx.x & 31;
    const int wave = threadIdx.x >> 5;      // column-tile 0..7
    const int l16  = lane & 15;
    const int hi   = lane >> 4;             // 0 or 1 (half-wave K-group select)
    const int row  = blockIdx.x * 16 + l16;
    const int col  = wave * 16 + l16;

    const _Float16* arow = A  + (size_t)row * K;   // A row (length K)
    const _Float16* bcol = WT + (size_t)col * K;   // column of W = row of WT (length K)

    v8f acc[2] = {{}, {}};
#pragma unroll
    for (int it = 0; it < K / 32; ++it) {
        const int kb = it * 32;
        // ISA 16-bit A 16x32 layout: lanes 0-15 hold K = kb+0..7 & kb+16..23,
        //                            lanes 16-31 hold K = kb+8..15 & kb+24..31.
        union { v16h v; v8h h[2]; } a, b;
        a.h[0] = *(const v8h*)(arow + kb + hi * 8);
        a.h[1] = *(const v8h*)(arow + kb + 16 + hi * 8);
        // B 32x16 layout (same K order as A): lanes 0-15 hold K = kb+0..7 & kb+16..23,
        // lanes 16-31 hold K = kb+8..15 & kb+24..31, all for column `col` of the WT row.
        b.h[0] = *(const v8h*)(bcol + kb + hi * 8);
        b.h[1] = *(const v8h*)(bcol + kb + 16 + hi * 8);
        acc[it & 1] = __builtin_amdgcn_wmma_f32_16x16x32_f16(
            /*neg_a=*/false, a.v, /*neg_b=*/false, b.v,
            /*c_mod=*/(short)0, acc[it & 1], /*reuse_a=*/false, /*reuse_b=*/false);
    }
    // C/D 16x16 f32 layout: VGPR v -> row (hi*8 + v), col = lane&15 within tile.
    float* crow = C + ((size_t)blockIdx.x * 16 + hi * 8) * DH + wave * 16 + l16;
#pragma unroll
    for (int v = 0; v < 8; ++v)
        crow[(size_t)v * DH] = acc[0][v] + acc[1][v];
}

// ---------------- self-loop + bias init of aggregate ----------------
__global__ void k_self(const float* __restrict__ zp, const float* __restrict__ zr,
                       const float* __restrict__ dp, const float* __restrict__ dr,
                       const float* __restrict__ bp, const float* __restrict__ br,
                       float* __restrict__ agg, int total) {
    int t = blockIdx.x * blockDim.x + threadIdx.x;
    if (t < total) {
        int i = t >> 7, d = t & 127;
        float wp = dp[i], wr = dr[i];
        agg[t] = zp[t] * wp * wp + bp[d] + zr[t] * wr * wr + br[d];
    }
}

// ---------------- fused edge scatter: both edge types, one edge per wave ----------------
// Wave index is forced uniform via readfirstlane so the per-edge src/dst/dinv loads
// can become scalar; the 128-float row gather stays a coalesced 512B/wave float4 read.
__global__ __launch_bounds__(256)
void k_edge2(const float* __restrict__ zp, const float* __restrict__ zr,
             const float* __restrict__ dip, const float* __restrict__ dir,
             const int* __restrict__ srcp, const int* __restrict__ dstp,
             const int* __restrict__ srcr, const int* __restrict__ dstr,
             float* __restrict__ agg) {
    const int lane = threadIdx.x & 31;
    unsigned wid = (unsigned)((blockIdx.x * blockDim.x + threadIdx.x) >> 5);
    wid = (unsigned)__builtin_amdgcn_readfirstlane((int)wid);   // wave-uniform scalar

    const float* z;
    const float* dinv;
    const int*   src;
    const int*   dst;
    unsigned e;
    if (wid < (unsigned)N_EDGES) {
        z = zp; dinv = dip; src = srcp; dst = dstp; e = wid;
    } else if (wid < 2u * N_EDGES) {
        z = zr; dinv = dir; src = srcr; dst = dstr; e = wid - N_EDGES;
    } else {
        return;
    }

    int s = src[e], d = dst[e];
    float w = dinv[s] * dinv[d];
    float4 zv = ((const float4*)(z + (size_t)s * DH))[lane];    // coalesced row gather
    float* o = agg + (size_t)d * DH + lane * 4;
    atomicAdd(o + 0, zv.x * w);
    atomicAdd(o + 1, zv.y * w);
    atomicAdd(o + 2, zv.z * w);
    atomicAdd(o + 3, zv.w * w);
}

// ---------------- ReLU in place + f16 cast for next layer ----------------
__global__ void k_relu_cast(float* __restrict__ h, _Float16* __restrict__ h16, int n) {
    int t = blockIdx.x * blockDim.x + threadIdx.x;
    if (t < n) {
        float v = h[t];
        v = v > 0.0f ? v : 0.0f;
        h[t] = v;
        h16[t] = (_Float16)v;
    }
}

// ---------------- pooling ----------------
__global__ void k_cnt(const int* __restrict__ batch, float* __restrict__ cnt) {
    int i = blockIdx.x * blockDim.x + threadIdx.x;
    if (i < N_NODES) atomicAdd(&cnt[batch[i]], 1.0f);
}

__global__ void k_pool(const float* __restrict__ h, const int* __restrict__ batch,
                       float* __restrict__ pooled) {
    int t = blockIdx.x * blockDim.x + threadIdx.x;
    if (t < N_NODES * DH) {
        int i = t >> 7, d = t & 127;
        atomicAdd(&pooled[(size_t)batch[i] * DH + d], h[t]);
    }
}

// ---------------- MLP head: one block (128 threads) per graph ----------------
__global__ __launch_bounds__(128)
void k_head(const float* __restrict__ pooled, const float* __restrict__ cnt,
            const float* __restrict__ l1W, const float* __restrict__ l1b,
            const float* __restrict__ l2W, const float* __restrict__ l2b,
            float* __restrict__ out) {
    __shared__ float mean[DH];
    __shared__ float red[DH];
    int g = blockIdx.x, t = threadIdx.x;
    float c = cnt[g];
    c = c < 1.0f ? 1.0f : c;
    mean[t] = pooled[(size_t)g * DH + t] / c;
    __syncthreads();
    float acc = l1b[t];
    for (int k = 0; k < DH; ++k) acc += mean[k] * l1W[(size_t)k * DH + t];
    float y = acc > 0.0f ? acc : 0.0f;
    red[t] = y * l2W[t];
    __syncthreads();
    for (int s = 64; s > 0; s >>= 1) {
        if (t < s) red[t] += red[t + s];
        __syncthreads();
    }
    if (t == 0) out[g] = red[0] + l2b[0];
}

// ---------------- launcher ----------------
extern "C" void kernel_launch(void* const* d_in, const int* in_sizes, int n_in,
                              void* d_out, int out_size, void* d_ws, size_t ws_size,
                              hipStream_t stream) {
    const float* x   = (const float*)d_in[0];
    const float* W0p = (const float*)d_in[1];
    const float* b0p = (const float*)d_in[2];
    const float* W0r = (const float*)d_in[3];
    const float* b0r = (const float*)d_in[4];
    const float* W1p = (const float*)d_in[5];
    const float* b1p = (const float*)d_in[6];
    const float* W1r = (const float*)d_in[7];
    const float* b1r = (const float*)d_in[8];
    const float* l1W = (const float*)d_in[9];
    const float* l1b = (const float*)d_in[10];
    const float* l2W = (const float*)d_in[11];
    const float* l2b = (const float*)d_in[12];
    const int* eip   = (const int*)d_in[13];   // [2, E]: row0 = src, row1 = dst
    const int* eir   = (const int*)d_in[14];
    const int* batch = (const int*)d_in[15];
    float* out = (float*)d_out;

    const int* src_p = eip, *dst_p = eip + N_EDGES;
    const int* src_r = eir, *dst_r = eir + N_EDGES;

    // carve workspace
    char* ws = (char*)d_ws;
    size_t off = 0;
    auto alloc = [&](size_t bytes) -> void* {
        void* p = ws + off;
        off += (bytes + 255) & ~(size_t)255;
        return p;
    };
    _Float16* xh   = (_Float16*)alloc((size_t)N_NODES * DIN * 2);
    _Float16* hh   = (_Float16*)alloc((size_t)N_NODES * DH * 2);
    float* zp      = (float*)alloc((size_t)N_NODES * DH * 4);
    float* zr      = (float*)alloc((size_t)N_NODES * DH * 4);
    float* agg     = (float*)alloc((size_t)N_NODES * DH * 4);
    float* dip     = (float*)alloc((size_t)N_NODES * 4);
    float* dir     = (float*)alloc((size_t)N_NODES * 4);
    _Float16* wt0p = (_Float16*)alloc((size_t)DH * DIN * 2);
    _Float16* wt0r = (_Float16*)alloc((size_t)DH * DIN * 2);
    _Float16* wt1p = (_Float16*)alloc((size_t)DH * DH * 2);
    _Float16* wt1r = (_Float16*)alloc((size_t)DH * DH * 2);
    float* pooled  = (float*)alloc((size_t)N_GRAPHS * DH * 4);
    float* cnt     = (float*)alloc((size_t)N_GRAPHS * 4);

    const int B = 256;
    auto gr = [](long n, int b) { return (int)((n + b - 1) / b); };

    // degrees -> dinv (deg init 1.0 covers the self loop)
    k_fill<<<gr(N_NODES, B), B, 0, stream>>>(dip, 1.0f, N_NODES);
    k_fill<<<gr(N_NODES, B), B, 0, stream>>>(dir, 1.0f, N_NODES);
    k_deg_count<<<gr(N_EDGES, B), B, 0, stream>>>(dst_p, dip, N_EDGES);
    k_deg_count<<<gr(N_EDGES, B), B, 0, stream>>>(dst_r, dir, N_EDGES);
    k_rsqrt<<<gr(N_NODES, B), B, 0, stream>>>(dip, N_NODES);
    k_rsqrt<<<gr(N_NODES, B), B, 0, stream>>>(dir, N_NODES);

    // precision prep: x -> f16, weights -> f16 transposed
    k_f32_to_f16<<<gr((long)N_NODES * DIN, B), B, 0, stream>>>(x, xh, N_NODES * DIN);
    k_wt<<<gr(DIN * DH, B), B, 0, stream>>>(W0p, wt0p, DIN);
    k_wt<<<gr(DIN * DH, B), B, 0, stream>>>(W0r, wt0r, DIN);
    k_wt<<<gr(DH * DH, B), B, 0, stream>>>(W1p, wt1p, DH);
    k_wt<<<gr(DH * DH, B), B, 0, stream>>>(W1r, wt1r, DH);

    const int rowTiles = N_NODES / 16;      // 3125, exact
    const int nElem = N_NODES * DH;
    const int edgeBlocks = gr((long)2 * N_EDGES * 32, B);   // both edge types in one launch

    // -------- layer 0 --------
    k_gemm<DIN><<<rowTiles, 256, 0, stream>>>(xh, wt0p, zp);
    k_gemm<DIN><<<rowTiles, 256, 0, stream>>>(xh, wt0r, zr);
    k_self<<<gr(nElem, B), B, 0, stream>>>(zp, zr, dip, dir, b0p, b0r, agg, nElem);
    k_edge2<<<edgeBlocks, B, 0, stream>>>(zp, zr, dip, dir, src_p, dst_p, src_r, dst_r, agg);
    k_relu_cast<<<gr(nElem, B), B, 0, stream>>>(agg, hh, nElem);

    // -------- layer 1 --------
    k_gemm<DH><<<rowTiles, 256, 0, stream>>>(hh, wt1p, zp);
    k_gemm<DH><<<rowTiles, 256, 0, stream>>>(hh, wt1r, zr);
    k_self<<<gr(nElem, B), B, 0, stream>>>(zp, zr, dip, dir, b1p, b1r, agg, nElem);
    k_edge2<<<edgeBlocks, B, 0, stream>>>(zp, zr, dip, dir, src_p, dst_p, src_r, dst_r, agg);
    k_relu_cast<<<gr(nElem, B), B, 0, stream>>>(agg, hh, nElem);

    // -------- global mean pool + head --------
    k_fill<<<gr(N_GRAPHS * DH, B), B, 0, stream>>>(pooled, 0.0f, N_GRAPHS * DH);
    k_fill<<<1, B, 0, stream>>>(cnt, 0.0f, N_GRAPHS);
    k_cnt<<<gr(N_NODES, B), B, 0, stream>>>(batch, cnt);
    k_pool<<<gr(nElem, B), B, 0, stream>>>(agg, batch, pooled);
    k_head<<<N_GRAPHS, 128, 0, stream>>>(pooled, cnt, l1W, l1b, l2W, l2b, out);
}
